// WaveNet_8289286881943
// MI455X (gfx1250) — compile-verified, matrix operand order checked by reading, not run on hardware
//
#include <hip/hip_runtime.h>

// ---------------------------------------------------------------------------
// Fused WaveNet forward for MI455X (gfx1250, wave32, WMMA 16x16x32 bf16).
//
// Each 256-thread workgroup (8 waves) owns one (batch, 128-wide time tile)
// and runs all 40 gated-residual layers with activations resident in LDS
// (bf16 [t][ch]) over an extended 256-column region (halo 64 >= receptive
// field 40). The 256-channel skip accumulator lives in WMMA f32 accumulator
// registers (16 x v8f = 128 VGPRs/wave) for the whole kernel.
//
// Weights are pre-converted/pre-swizzled into the wave32 WMMA A-operand lane
// layout by a pack kernel; the main kernel double-buffers each layer's 30 KB
// tile block in LDS via gfx1250 async copies (global_load_async_to_lds_b128,
// ASYNCcnt), prefetching layer l+1 while layer l computes. A-tile loads are
// software-pipelined so WMMAs wait on the oldest DS load only.
// ---------------------------------------------------------------------------

typedef __attribute__((ext_vector_type(16))) __bf16 v16bf;
typedef __attribute__((ext_vector_type(8)))  float  v8f;
typedef __attribute__((ext_vector_type(4)))  unsigned int u32x4;
typedef unsigned short u16;
typedef unsigned int   u32;

#define N_LAYERS   40
#define L_TOT      8192
#define BATCH      8
#define RC         32     // residual / dilation channels
#define SC         256    // skip channels
#define E_TILE     256    // extended columns per workgroup (16 N-tiles)
#define T_OUT      128    // valid output columns per workgroup
#define HALO       64
#define TILE_U16   512    // one A tile: 32 lanes x 16 bf16
#define LAYER_TILES 30    // 6 filter + 6 gate + 2 res + 16 skip
#define LAYER_STRIDE (LAYER_TILES * TILE_U16)      // 15360 u16 = 30 KB
#define END1_OFF   (N_LAYERS * LAYER_STRIDE)
#define WS_ELEMS   (END1_OFF + 8 * TILE_U16)       // + end1 (16x256)

__device__ __forceinline__ u16 f32_to_bf16(float f) {
  return __builtin_bit_cast(u16, static_cast<__bf16>(f));   // native cvt
}
__device__ __forceinline__ float bf16_to_f32(u16 h) {
  union { u32 u; float f; } v; v.u = ((u32)h) << 16;
  return v.f;
}
// ISA 16-bit A/B operand K pattern: element e holds
// K = (e<8 ? e : e+8) + (lane<16 ? 0 : 8).
__device__ __forceinline__ int kpat(int e, int lane) {
  return (e < 8 ? e : e + 8) + ((lane < 16) ? 0 : 8);
}
__device__ __forceinline__ v8f zerov8() {
  v8f z = {0.f, 0.f, 0.f, 0.f, 0.f, 0.f, 0.f, 0.f};
  return z;
}
__device__ __forceinline__ v8f wmma_bf16(v16bf a, v16bf b, v8f c) {
  return __builtin_amdgcn_wmma_f32_16x16x32_bf16(
      false, a, false, b, (short)0, c, false, false);
}
// Pre-swizzled A tile: 32 lanes x 16 bf16, lane-major (32 B/lane).
__device__ __forceinline__ v16bf ldtile(const u16* p, int lane) {
  return reinterpret_cast<const v16bf*>(p)[lane];           // 2x ds_load_b128
}
// 32x16 (KxN) B operand from a channel-contiguous LDS row. The K pattern is
// two contiguous 16-byte runs per lane -> 2x ds_load_b128.
__device__ __forceinline__ v16bf lds_B(const u16* colbase, int laneHi) {
  union { v16bf v; u32x4 q[2]; } r;
  const u32x4* p = reinterpret_cast<const u32x4*>(colbase + laneHi);
  r.q[0] = p[0];        // K = hi+0..7   (bytes +0..15)
  r.q[1] = p[2];        // K = hi+16..23 (bytes +32..47)
  return r.v;
}
// gfx1250 async copy: per-lane 16 B global -> LDS, tracked on ASYNCcnt.
__device__ __forceinline__ void async_g2l_b128(u32 lds_off, const u16* gptr) {
  asm volatile("global_load_async_to_lds_b128 %0, %1, off"
               :: "v"(lds_off), "v"((unsigned long long)(size_t)gptr)
               : "memory");
}
__device__ __forceinline__ void wait_async0() {
  asm volatile("s_wait_asynccnt 0x0" ::: "memory");
}
__device__ __forceinline__ void stage_layer_async(u16* dst_lds,
                                                  const u16* src, int tid) {
  u32 base = (u32)(size_t)(void*)dst_lds;       // low 32 bits = LDS offset
  for (int i = tid; i < LAYER_STRIDE / 8; i += 256)   // 1920 x 16 B
    async_g2l_b128(base + (u32)(i * 16), src + i * 8);
}

// ---------------------------------------------------------------------------
// Pack kernel: f32 weights -> bf16 tiles in WMMA A-operand lane layout.
// Per layer: filter[mt][kc] (6), gate (6), res[mt] (2), skip[mt] (16).
// Filter/gate GEMM K index = tap*32 + c_in so B K-chunk kc is the h row
// shifted by (kc-1).
// ---------------------------------------------------------------------------
__global__ void pack_weights(const float* __restrict__ filter_w,
                             const float* __restrict__ gate_w,
                             const float* __restrict__ res_w,
                             const float* __restrict__ skip_w,
                             const float* __restrict__ end1_w,
                             u16* __restrict__ wt) {
  int idx = blockIdx.x * blockDim.x + threadIdx.x;
  if (idx >= WS_ELEMS) return;
  int e      = idx & 15;
  int lane   = (idx >> 4) & 31;
  int tileid = idx >> 9;
  int m  = lane & 15;
  int kp = kpat(e, lane);
  float v;
  if (tileid < N_LAYERS * LAYER_TILES) {
    int l = tileid / LAYER_TILES;
    int t = tileid % LAYER_TILES;
    if (t < 6) {
      int mt = t / 3, kc = t % 3;
      v = filter_w[((l * 32 + mt * 16 + m) * 32 + kp) * 3 + kc];
    } else if (t < 12) {
      int tt = t - 6; int mt = tt / 3, kc = tt % 3;
      v = gate_w[((l * 32 + mt * 16 + m) * 32 + kp) * 3 + kc];
    } else if (t < 14) {
      int mt = t - 12;
      v = res_w[(l * 32 + mt * 16 + m) * 32 + kp];
    } else {
      int mt = t - 14;
      v = skip_w[(l * 256 + mt * 16 + m) * 32 + kp];
    }
  } else {
    int kc = tileid - N_LAYERS * LAYER_TILES;
    v = end1_w[m * 256 + kc * 32 + kp];
  }
  wt[idx] = f32_to_bf16(v);
}

// ---------------------------------------------------------------------------
// Main fused kernel.
// ---------------------------------------------------------------------------
__global__ __launch_bounds__(256)
void wavenet_fused(const float* __restrict__ x,
                   const float* __restrict__ start_w,
                   const u16*   __restrict__ wt,
                   const float* __restrict__ end1_b,
                   const float* __restrict__ end2_w,
                   const float* __restrict__ end2_b,
                   float* __restrict__ out) {
  // 48 KB activations (h ping/pong, z) + 2 x 30 KB weight staging buffers.
  __shared__ __align__(64) u16 lds[3 * E_TILE * RC];
  __shared__ __align__(64) u16 wlds[2][LAYER_STRIDE];
  u16* hbuf0 = lds;
  u16* hbuf1 = lds + E_TILE * RC;
  u16* zbuf  = lds + 2 * E_TILE * RC;

  const int b    = blockIdx.x / (L_TOT / T_OUT);
  const int tile = blockIdx.x % (L_TOT / T_OUT);
  const int t0   = tile * T_OUT - HALO;
  const int tid  = threadIdx.x;
  const int lane = tid & 31;
  const int wave = tid >> 5;
  const int nlo  = lane & 15;
  const int khi  = (lane < 16) ? 0 : 8;     // B K offset for hi half-wave
  const int Mofs = (lane < 16) ? 0 : 8;     // C/D row offset for hi half-wave

  // Prefetch layer 0 weights while we build h0.
  stage_layer_async(wlds[0], wt, tid);

  // h0 = start 1x1 conv of x (zero-extended outside the sequence).
  for (int i = tid; i < E_TILE * RC; i += 256) {
    int t = i >> 5, c = i & 31;
    int tg = t0 + t;
    float xv = (tg >= 0 && tg < L_TOT) ? x[b * L_TOT + tg] : 0.f;
    hbuf0[t * RC + c] = f32_to_bf16(xv * start_w[c]);
  }
  wait_async0();
  __syncthreads();

  v8f skip[16];
#pragma unroll
  for (int m = 0; m < 16; ++m) skip[m] = zerov8();

  u16* h_in  = hbuf0;
  u16* h_out = hbuf1;

#pragma unroll 1
  for (int l = 0; l < N_LAYERS; ++l) {
    const u16* wl = wlds[l & 1];
    // Async prefetch of next layer's tiles into the other buffer (its last
    // consumer finished at the previous end-of-layer barrier).
    if (l + 1 < N_LAYERS)
      stage_layer_async(wlds[(l + 1) & 1], wt + (l + 1) * LAYER_STRIDE, tid);

    // --- filter & gate k=3 convs; A tiles shared across both N-tiles ------
    v8f facc[2][2], gacc[2][2];                  // [s][mt]
#pragma unroll
    for (int s = 0; s < 2; ++s)
#pragma unroll
      for (int mt = 0; mt < 2; ++mt) {
        facc[s][mt] = zerov8();
        gacc[s][mt] = zerov8();
      }
#pragma unroll
    for (int kc = 0; kc < 3; ++kc) {             // taps t-1, t, t+1
      // Group-preload the 4 A tiles of this tap, then both B rows.
      v16bf Af0 = ldtile(wl + (kc)     * TILE_U16, lane);
      v16bf Ag0 = ldtile(wl + (6 + kc) * TILE_U16, lane);
      v16bf Af1 = ldtile(wl + (3 + kc) * TILE_U16, lane);
      v16bf Ag1 = ldtile(wl + (9 + kc) * TILE_U16, lane);
#pragma unroll
      for (int s = 0; s < 2; ++s) {
        int trow = (wave * 2 + s) * 16 + nlo + kc - 1;
        trow = trow < 0 ? 0 : (trow > E_TILE - 1 ? E_TILE - 1 : trow);
        v16bf Bt = lds_B(h_in + trow * RC, khi);
        facc[s][0] = wmma_bf16(Af0, Bt, facc[s][0]);
        gacc[s][0] = wmma_bf16(Ag0, Bt, gacc[s][0]);
        facc[s][1] = wmma_bf16(Af1, Bt, facc[s][1]);
        gacc[s][1] = wmma_bf16(Ag1, Bt, gacc[s][1]);
      }
    }
    // z = tanh(f) * sigmoid(g) via fast v_exp/v_rcp; packed 16 B DS stores.
#pragma unroll
    for (int s = 0; s < 2; ++s) {
      const int tcol = (wave * 2 + s) * 16 + nlo;
#pragma unroll
      for (int mt = 0; mt < 2; ++mt) {
        union { u32x4 q; u16 h[8]; } o;
#pragma unroll
        for (int e = 0; e < 8; ++e) {
          float fv = facc[s][mt][e], gv = gacc[s][mt][e];
          float th = 1.f - 2.f * __builtin_amdgcn_rcpf(__expf(2.f * fv) + 1.f);
          float sg = __builtin_amdgcn_rcpf(1.f + __expf(-gv));
          o.h[e] = f32_to_bf16(th * sg);
        }
        *reinterpret_cast<u32x4*>(&zbuf[tcol * RC + mt * 16 + Mofs]) = o.q;
      }
    }
    __syncthreads();   // z complete across all waves

    // --- skip 1x1 (32->256), software-pipelined A-tile loads (depth 2) ----
    {
      v16bf Bz = lds_B(zbuf + (HALO + wave * 16 + nlo) * RC, khi);
      v16bf a0 = ldtile(wl + 14 * TILE_U16, lane);
      v16bf a1 = ldtile(wl + 15 * TILE_U16, lane);
#pragma unroll
      for (int mt = 0; mt < 16; ++mt) {
        v16bf an = a0;
        if (mt < 14) an = ldtile(wl + (16 + mt) * TILE_U16, lane);
        skip[mt] = wmma_bf16(a0, Bz, skip[mt]);   // consumes oldest load
        a0 = a1; a1 = an;
      }
    }

    // --- residual 1x1 (32->32) + add; res A tiles shared across N-tiles ---
    {
      v16bf Ar0 = ldtile(wl + 12 * TILE_U16, lane);
      v16bf Ar1 = ldtile(wl + 13 * TILE_U16, lane);
#pragma unroll
      for (int s = 0; s < 2; ++s) {
        const int tcol = (wave * 2 + s) * 16 + nlo;
        v16bf Bz = lds_B(zbuf + tcol * RC, khi);
        union { u32x4 q; u16 h[8]; } hi0, hi1, ho;
        hi0.q = *reinterpret_cast<const u32x4*>(&h_in[tcol * RC + Mofs]);
        hi1.q = *reinterpret_cast<const u32x4*>(&h_in[tcol * RC + 16 + Mofs]);
        v8f r0 = wmma_bf16(Ar0, Bz, zerov8());
        v8f r1 = wmma_bf16(Ar1, Bz, zerov8());
#pragma unroll
        for (int e = 0; e < 8; ++e)
          ho.h[e] = f32_to_bf16(r0[e] + bf16_to_f32(hi0.h[e]));
        *reinterpret_cast<u32x4*>(&h_out[tcol * RC + Mofs]) = ho.q;
#pragma unroll
        for (int e = 0; e < 8; ++e)
          ho.h[e] = f32_to_bf16(r1[e] + bf16_to_f32(hi1.h[e]));
        *reinterpret_cast<u32x4*>(&h_out[tcol * RC + 16 + Mofs]) = ho.q;
      }
    }
    wait_async0();     // next layer's weights have landed in LDS
    __syncthreads();   // h_out / weight buffer visible to all waves

    u16* tmp = h_in; h_in = h_out; h_out = tmp;
  }

  // ---- end stage: relu -> 1x1 (256->16) + bias -> relu -> 1x1 (16->1) ----
  u16*   stg = lds + wave * (RC * 16);                            // 1 KB/wave
  float* o1  = reinterpret_cast<float*>(lds + 8 * RC * 16) + wave * 256;

  v8f e1 = zerov8();
#pragma unroll
  for (int kc = 0; kc < 8; ++kc) {
    // Re-stage one 32-channel K-chunk of relu(skip) into B layout (16 B DS).
#pragma unroll
    for (int half = 0; half < 2; ++half) {
      int mt = 2 * kc + half;
      union { u32x4 q; u16 h[8]; } o;
#pragma unroll
      for (int e = 0; e < 8; ++e) {
        float sv = skip[mt][e];
        o.h[e] = f32_to_bf16(sv > 0.f ? sv : 0.f);
      }
      *reinterpret_cast<u32x4*>(&stg[nlo * RC + half * 16 + Mofs]) = o.q;
    }
    v16bf Bs = lds_B(stg + nlo * RC, khi);     // intra-wave; DScnt ordered
    e1 = wmma_bf16(ldtile(wt + END1_OFF + kc * TILE_U16, lane), Bs, e1);
  }
#pragma unroll
  for (int e = 0; e < 8; ++e) {
    int M = e + Mofs;
    float v = e1[e] + end1_b[M];
    o1[M * 16 + nlo] = v > 0.f ? v : 0.f;
  }
  if (lane < 16) {
    float acc = end2_b[0];
#pragma unroll
    for (int e2 = 0; e2 < 16; ++e2) acc += end2_w[e2] * o1[e2 * 16 + lane];
    out[b * L_TOT + tile * T_OUT + wave * 16 + lane] = acc;
  }
}

// ---------------------------------------------------------------------------
extern "C" void kernel_launch(void* const* d_in, const int* in_sizes, int n_in,
                              void* d_out, int out_size, void* d_ws, size_t ws_size,
                              hipStream_t stream) {
  (void)in_sizes; (void)n_in; (void)out_size; (void)ws_size;
  const float* x        = (const float*)d_in[0];
  const float* start_w  = (const float*)d_in[1];
  const float* filter_w = (const float*)d_in[2];
  const float* gate_w   = (const float*)d_in[3];
  const float* res_w    = (const float*)d_in[4];
  const float* skip_w   = (const float*)d_in[5];
  const float* end1_w   = (const float*)d_in[6];
  const float* end1_b   = (const float*)d_in[7];
  const float* end2_w   = (const float*)d_in[8];
  const float* end2_b   = (const float*)d_in[9];
  u16*   wt  = (u16*)d_ws;              // ~1.21 MB pre-swizzled bf16 tiles
  float* out = (float*)d_out;

  pack_weights<<<(WS_ELEMS + 255) / 256, 256, 0, stream>>>(
      filter_w, gate_w, res_w, skip_w, end1_w, wt);

  wavenet_fused<<<BATCH * (L_TOT / T_OUT), 256, 0, stream>>>(
      x, start_w, wt, end1_b, end2_w, end2_b, out);
}
